// VSSBlock_CrossScan_23699629539953
// MI455X (gfx1250) — compile-verified, matrix operand order checked by reading, not run on hardware
//
#include <hip/hip_runtime.h>
#include <stdint.h>

typedef __bf16 bf16_t;
typedef __attribute__((ext_vector_type(8)))  __bf16 v8bf;
typedef __attribute__((ext_vector_type(16))) __bf16 v16bf;
typedef __attribute__((ext_vector_type(8)))  float  v8f;
typedef __attribute__((ext_vector_type(4)))  float  v4f;

// Problem constants
constexpr int kDM    = 192;    // model dim
constexpr int kDIN   = 384;    // inner dim
constexpr int kNH    = 6;      // heads
constexpr int kHD    = 64;     // head dim
constexpr int kDC    = 4;      // conv width
constexpr int kBB    = 2;
constexpr int kHH    = 64;
constexpr int kWW    = 64;
constexpr int kLSEQ  = kHH * kWW;          // 4096
constexpr int kNSEQ  = 4 * kBB;            // 8 scan sequences
constexpr int kMROWS = kNSEQ * kLSEQ;      // 32768
constexpr int kDPROJ = 2 * kDIN + 2 * 128 + kNH; // 1030
constexpr int kCONVD = kDIN + 2 * 128;     // 640
constexpr int kNCHUNK = kLSEQ / 64;        // 64 chunks of 64
constexpr int kPIX   = kBB * kHH * kWW;    // 8192

__host__ __device__ constexpr int pad32(int n) { return (n + 31) & ~31; }

// ---------------------------------------------------------------------------
// WMMA fragment load, 16-bit A/B layout (cdna5 §7.12.2):
// lane-half selects K base (0 or 8); elements 0-7 -> K+base+0..7 (16B run),
// elements 8-15 -> K+base+16..23 (16B run).  Two b128 loads per fragment.
__device__ __forceinline__ v16bf load_frag(const bf16_t* rowk, int lane) {
  const v8bf* p = (const v8bf*)(rowk + ((lane >> 4) << 3));
  const v8bf lo = p[0];
  const v8bf hi = p[2];
  return __builtin_shufflevector(lo, hi, 0, 1, 2, 3, 4, 5, 6, 7,
                                 8, 9, 10, 11, 12, 13, 14, 15);
}

__device__ __forceinline__ v8f wmma_bf16(v16bf a, v16bf b, v8f c) {
  return __builtin_amdgcn_wmma_f32_16x16x32_bf16(false, a, false, b,
                                                 (short)0, c, false, false);
}

// ---------------------------------------------------------------------------
// Generic bf16 WMMA GEMM:  out[M,N] = A[M,K] @ W[Npad,K]^T  (+bias/gelu/resid)
// W is zero-padded to pad32(N) rows so fragment loads are unconditional.
// Each wave computes a 16x32 (two 16x16) output tile sharing one A fragment.
#define EPI_BIAS   1
#define EPI_GELU   2
#define EPI_RESID  4
#define EPI_BF16   8

__global__ void gemm_wmma_bf16(const bf16_t* __restrict__ A,
                               const bf16_t* __restrict__ W,
                               const float* __restrict__ bias,
                               const float* __restrict__ resid,
                               float* __restrict__ outF,
                               bf16_t* __restrict__ outB,
                               int M, int N, int K, int mode) {
  const int lane = threadIdx.x & 31;
  const int wid  = threadIdx.x >> 5;
  const int tiles_n = (N + 15) >> 4;
  const int pairs_n = (tiles_n + 1) >> 1;
  const int pair = blockIdx.x * 8 + wid;
  const int total = (M >> 4) * pairs_n;
  if (pair >= total) return;                 // wave-uniform
  const int tm = (pair / pairs_n) << 4;
  const int tn = (pair % pairs_n) << 5;
  const bf16_t* arow  = A + (size_t)(tm + (lane & 15)) * K;
  const bf16_t* wrow0 = W + (size_t)(tn + (lane & 15)) * K;
  const bf16_t* wrow1 = W + (size_t)(tn + 16 + (lane & 15)) * K;

  v8f acc0 = {}, acc1 = {};
  for (int k0 = 0; k0 < K; k0 += 32) {
    __builtin_prefetch(arow + k0 + 32, 0, 3);
    const v16bf af  = load_frag(arow + k0, lane);
    const v16bf b0  = load_frag(wrow0 + k0, lane);
    const v16bf b1  = load_frag(wrow1 + k0, lane);
    acc0 = wmma_bf16(af, b0, acc0);
    acc1 = wmma_bf16(af, b1, acc1);
  }
#pragma unroll
  for (int tile = 0; tile < 2; ++tile) {
    const v8f acc = tile ? acc1 : acc0;
    const int tnn = tn + tile * 16;
#pragma unroll
    for (int r = 0; r < 8; ++r) {
      const int m = tm + r + ((lane >> 4) << 3);
      const int n = tnn + (lane & 15);
      if (n >= N) continue;
      float v = acc[r];
      if (mode & EPI_BIAS)  v += bias[n];
      if (mode & EPI_GELU)  v = 0.5f * v * (1.0f + erff(v * 0.70710678f));
      if (mode & EPI_RESID) v += resid[(size_t)m * N + n];
      if (mode & EPI_BF16)  outB[(size_t)m * N + n] = (bf16_t)v;
      else                  outF[(size_t)m * N + n] = v;
    }
  }
}

// ---------------------------------------------------------------------------
// LayerNorm over 192 channels + 4-direction cross-scan, bf16 output.
__global__ void ln_crossscan(const float* __restrict__ x,
                             const float* __restrict__ w,
                             const float* __restrict__ b,
                             bf16_t* __restrict__ xs) {
  const int pix = blockIdx.x;                       // b*4096 + hh*64 + ww
  const int bb = pix >> 12, hh = (pix >> 6) & 63, ww = pix & 63;
  const int tid = threadIdx.x;
  __shared__ float red[256];
  const float* row = x + (size_t)pix * kDM;
  const float v = (tid < kDM) ? row[tid] : 0.0f;
  red[tid] = v; __syncthreads();
  for (int s = 128; s > 0; s >>= 1) { if (tid < s) red[tid] += red[tid + s]; __syncthreads(); }
  const float mu = red[0] / kDM; __syncthreads();
  const float d = (tid < kDM) ? (v - mu) : 0.0f;
  red[tid] = d * d; __syncthreads();
  for (int s = 128; s > 0; s >>= 1) { if (tid < s) red[tid] += red[tid + s]; __syncthreads(); }
  const float var = red[0] / kDM;
  if (tid < kDM) {
    const bf16_t h16 = (bf16_t)((v - mu) * rsqrtf(var + 1e-5f) * w[tid] + b[tid]);
    const int r0 = (0 * kBB + bb) * kLSEQ + hh * kWW + ww;
    const int r1 = (1 * kBB + bb) * kLSEQ + hh * kWW + (kWW - 1 - ww);
    const int r2 = (2 * kBB + bb) * kLSEQ + ww * kHH + hh;
    const int r3 = (3 * kBB + bb) * kLSEQ + ww * kHH + (kHH - 1 - hh);
    xs[(size_t)r0 * kDM + tid] = h16;
    xs[(size_t)r1 * kDM + tid] = h16;
    xs[(size_t)r2 * kDM + tid] = h16;
    xs[(size_t)r3 * kDM + tid] = h16;
  }
}

// ---------------------------------------------------------------------------
// Depthwise conv(4) + SiLU + dt softplus; emit xh(f32), X=xh*dt, B, C (bf16), dA.
__global__ void conv_gate(const float* __restrict__ zx,
                          const float* __restrict__ cw,
                          const float* __restrict__ cb,
                          const float* __restrict__ dt_bias,
                          const float* __restrict__ A_log,
                          float* __restrict__ xh,
                          bf16_t* __restrict__ Xb,
                          bf16_t* __restrict__ Bb,
                          bf16_t* __restrict__ Cb,
                          float* __restrict__ dA) {
  const int row = blockIdx.x;                 // s*4096 + t
  const int s = row >> 12, t = row & 4095;
  const int tid = threadIdx.x;
  __shared__ float dt_sh[kNH];
  if (tid < kNH) {
    const float raw = zx[(size_t)row * kDPROJ + (kDPROJ - kNH) + tid] + dt_bias[tid];
    const float sp = (raw > 20.0f) ? raw : log1pf(expf(raw));
    dt_sh[tid] = sp;
    dA[(size_t)row * kNH + tid] = sp * (-expf(A_log[tid]));
  }
  __syncthreads();
  for (int c = tid; c < kCONVD; c += blockDim.x) {
    float a = cb[c];
#pragma unroll
    for (int k = 0; k < kDC; ++k) {
      const int tt = t - (kDC - 1) + k;
      if (tt >= 0)
        a += zx[((size_t)(s * kLSEQ + tt)) * kDPROJ + kDIN + c] * cw[c * kDC + k];
    }
    const float v = a / (1.0f + expf(-a));    // silu
    if (c < kDIN) {
      xh[(size_t)row * kDIN + c] = v;
      const int h = c >> 6;
      Xb[((size_t)row * kNH + h) * kHD + (c & 63)] = (bf16_t)(v * dt_sh[h]);
    } else if (c < kDIN + 128) {
      Bb[(size_t)row * 128 + (c - kDIN)] = (bf16_t)v;
    } else {
      Cb[(size_t)row * 128 + (c - kDIN - 128)] = (bf16_t)v;
    }
  }
}

// ---------------------------------------------------------------------------
// SSD intra-chunk: G=C@B^T, L-mask, Ydiag=L@X, states=X^T@(B*decay).
// LDS layouts chosen so every WMMA fragment read is a contiguous b128:
//   Csh[t][n], Bsh[t][n]   -> G fragments
//   Xt[p][t]               -> Ydiag B-fragment and states A-fragment
//   Btd[n][t] (=B*decay^T) -> states B-fragment
//   Lsh[s][t]              -> aliases Csh after G completes
__global__ void ssd_intra(const bf16_t* __restrict__ Bb,
                          const bf16_t* __restrict__ Cb,
                          const bf16_t* __restrict__ Xb,
                          const float* __restrict__ dA,
                          float* __restrict__ AcumBuf,
                          float* __restrict__ cdec,
                          float* __restrict__ states,
                          float* __restrict__ Ybuf) {
  const int c = blockIdx.x, h = blockIdx.y, s = blockIdx.z;
  const int tid = threadIdx.x, lane = tid & 31, wid = tid >> 5;
  __shared__ __align__(16) char smem[56 * 1024 + 256];
  bf16_t (*Csh)[128] = reinterpret_cast<bf16_t(*)[128]>(smem);
  bf16_t (*Bsh)[128] = reinterpret_cast<bf16_t(*)[128]>(smem + 16 * 1024);
  bf16_t (*Btd)[64]  = reinterpret_cast<bf16_t(*)[64]>(smem + 32 * 1024);
  bf16_t (*Xt)[64]   = reinterpret_cast<bf16_t(*)[64]>(smem + 48 * 1024);
  bf16_t (*Lsh)[64]  = reinterpret_cast<bf16_t(*)[64]>(smem);   // alias Csh
  float* Ac = reinterpret_cast<float*>(smem + 56 * 1024);

  if (tid == 0) {
    float a = 0.0f;
    for (int i = 0; i < 64; ++i) {
      a += dA[((size_t)(s * kLSEQ) + c * 64 + i) * kNH + h];
      Ac[i] = a;
      AcumBuf[((size_t)(s * kNH + h)) * kLSEQ + c * 64 + i] = a;
    }
    cdec[(s * kNH + h) * kNCHUNK + c] = expf(a);
  }
  __syncthreads();
  const float alast = Ac[63];
  const size_t rowbase = (size_t)(s * kLSEQ) + c * 64;
  for (int idx = tid; idx < 64 * 16; idx += 256) {
    const int t = idx >> 4, ch = idx & 15;
    const v8bf bv = *(const v8bf*)(Bb + (rowbase + t) * 128 + ch * 8);
    const v8bf cv = *(const v8bf*)(Cb + (rowbase + t) * 128 + ch * 8);
    *(v8bf*)(&Bsh[t][ch * 8]) = bv;
    *(v8bf*)(&Csh[t][ch * 8]) = cv;
    const float dec = expf(alast - Ac[t]);
#pragma unroll
    for (int j = 0; j < 8; ++j)
      Btd[ch * 8 + j][t] = (bf16_t)((float)bv[j] * dec);
  }
  for (int idx = tid; idx < 64 * 8; idx += 256) {
    const int t = idx >> 3, ch = idx & 7;
    const v8bf xv = *(const v8bf*)(Xb + ((rowbase + t) * kNH + h) * kHD + ch * 8);
#pragma unroll
    for (int j = 0; j < 8; ++j) Xt[ch * 8 + j][t] = xv[j];
  }
  __syncthreads();

  // G = C @ B^T (64x64, K=128); keep masked L values in registers.
  float lv[2][8];
#pragma unroll
  for (int it = 0; it < 2; ++it) {
    const int tile = wid * 2 + it;
    const int m0 = (tile >> 2) << 4;            // query index
    const int n0 = (tile & 3) << 4;             // key index
    v8f acc = {};
    for (int k0 = 0; k0 < 128; k0 += 32)
      acc = wmma_bf16(load_frag(&Csh[m0 + (lane & 15)][k0], lane),
                      load_frag(&Bsh[n0 + (lane & 15)][k0], lane), acc);
#pragma unroll
    for (int r = 0; r < 8; ++r) {
      const int sr = m0 + r + ((lane >> 4) << 3);
      const int tc = n0 + (lane & 15);
      lv[it][r] = (sr >= tc) ? acc[r] * expf(Ac[sr] - Ac[tc]) : 0.0f;
    }
  }
  __syncthreads();                              // all G reads of Csh done
#pragma unroll
  for (int it = 0; it < 2; ++it) {
    const int tile = wid * 2 + it;
    const int m0 = (tile >> 2) << 4;
    const int n0 = (tile & 3) << 4;
#pragma unroll
    for (int r = 0; r < 8; ++r)
      Lsh[m0 + r + ((lane >> 4) << 3)][n0 + (lane & 15)] = (bf16_t)lv[it][r];
  }
  __syncthreads();

  // Ydiag = L @ X (64x64, K=64): A=Lsh rows, B=Xt rows (both contiguous).
#pragma unroll
  for (int it = 0; it < 2; ++it) {
    const int tile = wid * 2 + it;
    const int m0 = (tile >> 2) << 4;
    const int n0 = (tile & 3) << 4;
    v8f acc = {};
    for (int k0 = 0; k0 < 64; k0 += 32)
      acc = wmma_bf16(load_frag(&Lsh[m0 + (lane & 15)][k0], lane),
                      load_frag(&Xt[n0 + (lane & 15)][k0], lane), acc);
#pragma unroll
    for (int r = 0; r < 8; ++r) {
      const int tr = m0 + r + ((lane >> 4) << 3);
      const int p  = n0 + (lane & 15);
      Ybuf[(((size_t)rowbase + tr) * kNH + h) * kHD + p] = acc[r];
    }
  }

  // states[p][n] = sum_t X[t][p] * B[t][n] * dec[t]: A=Xt rows, B=Btd rows.
#pragma unroll
  for (int it = 0; it < 4; ++it) {
    const int tile = wid * 4 + it;
    const int m0 = (tile >> 3) << 4;            // p
    const int n0 = (tile & 7) << 4;             // n
    v8f acc = {};
    for (int k0 = 0; k0 < 64; k0 += 32)
      acc = wmma_bf16(load_frag(&Xt[m0 + (lane & 15)][k0], lane),
                      load_frag(&Btd[n0 + (lane & 15)][k0], lane), acc);
#pragma unroll
    for (int r = 0; r < 8; ++r) {
      const int p = m0 + r + ((lane >> 4) << 3);
      const int n = n0 + (lane & 15);
      states[((((size_t)(s * kNH + h)) * kNCHUNK + c) * 64 + p) * 128 + n] = acc[r];
    }
  }
}

// ---------------------------------------------------------------------------
// Sequential inter-chunk recurrence (float4); overwrites states with the
// EXCLUSIVE prefix (prev state entering each chunk).
__global__ void ssd_scan(float* __restrict__ states,
                         const float* __restrict__ cdec) {
  const int sh = blockIdx.x;                  // (s*NH+h), 48 blocks
  const int tid = threadIdx.x;                // 256
  v4f carry[8];
#pragma unroll
  for (int j = 0; j < 8; ++j) carry[j] = {};
  float* base = states + (size_t)sh * kNCHUNK * 8192;
  const float* dec = cdec + sh * kNCHUNK;
  for (int c = 0; c < kNCHUNK; ++c) {
    const float d = dec[c];
    v4f* sc = (v4f*)(base + (size_t)c * 8192);
#pragma unroll
    for (int j = 0; j < 8; ++j) {
      const int idx = tid + j * 256;
      const v4f st = sc[idx];
      const v4f pv = carry[j];
      carry[j] = pv * d + st;
      sc[idx] = pv;
    }
  }
}

// ---------------------------------------------------------------------------
// SSD inter-chunk: Yoff = (C * exp(Acum)) @ prev^T, accumulated into Ybuf.
__global__ void ssd_inter(const bf16_t* __restrict__ Cb,
                          const float* __restrict__ AcumBuf,
                          const float* __restrict__ states,
                          float* __restrict__ Ybuf) {
  const int c = blockIdx.x, h = blockIdx.y, s = blockIdx.z;
  const int tid = threadIdx.x, lane = tid & 31, wid = tid >> 5;
  __shared__ __align__(16) char smem[32 * 1024];
  bf16_t (*Cex)[128] = reinterpret_cast<bf16_t(*)[128]>(smem);
  bf16_t (*Psh)[128] = reinterpret_cast<bf16_t(*)[128]>(smem + 16 * 1024);
  const size_t rowbase = (size_t)(s * kLSEQ) + c * 64;
  const size_t stbase = (((size_t)(s * kNH + h)) * kNCHUNK + c) * 64 * 128;
  for (int idx = tid; idx < 64 * 16; idx += 256) {
    const int t = idx >> 4, ch = idx & 15;
    const float ea = expf(AcumBuf[((size_t)(s * kNH + h)) * kLSEQ + c * 64 + t]);
    const v8bf cv = *(const v8bf*)(Cb + (rowbase + t) * 128 + ch * 8);
    const v4f p0 = *(const v4f*)(states + stbase + t * 128 + ch * 8);
    const v4f p1 = *(const v4f*)(states + stbase + t * 128 + ch * 8 + 4);
    v8bf ce, pe;
#pragma unroll
    for (int j = 0; j < 4; ++j) {
      ce[j] = (bf16_t)((float)cv[j] * ea);
      ce[j + 4] = (bf16_t)((float)cv[j + 4] * ea);
      pe[j] = (bf16_t)p0[j];
      pe[j + 4] = (bf16_t)p1[j];
    }
    *(v8bf*)(&Cex[t][ch * 8]) = ce;
    *(v8bf*)(&Psh[t][ch * 8]) = pe;             // Psh[p][n]
  }
  __syncthreads();
#pragma unroll
  for (int it = 0; it < 2; ++it) {
    const int tile = wid * 2 + it;
    const int m0 = (tile >> 2) << 4;            // t
    const int n0 = (tile & 3) << 4;             // p
    v8f acc = {};
    for (int k0 = 0; k0 < 128; k0 += 32)
      acc = wmma_bf16(load_frag(&Cex[m0 + (lane & 15)][k0], lane),
                      load_frag(&Psh[n0 + (lane & 15)][k0], lane), acc);
#pragma unroll
    for (int r = 0; r < 8; ++r) {
      const int tr = m0 + r + ((lane >> 4) << 3);
      const int p  = n0 + (lane & 15);
      Ybuf[(((size_t)rowbase + tr) * kNH + h) * kHD + p] += acc[r];
    }
  }
}

// ---------------------------------------------------------------------------
// y = (Y + D*xh) * silu(z); RMS-norm over 384; bf16 out for out_proj.
__global__ void gate_rms(const float* __restrict__ Ybuf,
                         const float* __restrict__ xh,
                         const float* __restrict__ zx,
                         const float* __restrict__ Dh,
                         const float* __restrict__ rms_w,
                         bf16_t* __restrict__ yg) {
  const int row = blockIdx.x;
  const int tid = threadIdx.x;                // 128 threads x 3 channels
  __shared__ float red[128];
  float vals[3];
  float ss = 0.0f;
#pragma unroll
  for (int j = 0; j < 3; ++j) {
    const int d = tid + j * 128;
    const int h = d >> 6;
    float y = Ybuf[((size_t)row * kNH + h) * kHD + (d & 63)] +
              Dh[h] * xh[(size_t)row * kDIN + d];
    const float z = zx[(size_t)row * kDPROJ + d];
    y *= z / (1.0f + expf(-z));
    vals[j] = y; ss += y * y;
  }
  red[tid] = ss; __syncthreads();
  for (int s = 64; s > 0; s >>= 1) { if (tid < s) red[tid] += red[tid + s]; __syncthreads(); }
  const float inv = rsqrtf(red[0] / kDIN + 1e-5f);
#pragma unroll
  for (int j = 0; j < 3; ++j) {
    const int d = tid + j * 128;
    yg[(size_t)row * kDIN + d] = (bf16_t)(vals[j] * inv * rms_w[d]);
  }
}

// ---------------------------------------------------------------------------
// Cross-merge (undo the 4 scan orders) + residual.
__global__ void cross_merge(const float* __restrict__ x,
                            const float* __restrict__ yp,
                            float* __restrict__ x1) {
  const int idx = blockIdx.x * blockDim.x + threadIdx.x;
  if (idx >= kPIX * kDM) return;
  const int cch = idx % kDM;
  const int ww  = (idx / kDM) % kWW;
  const int hh  = (idx / (kDM * kWW)) % kHH;
  const int bb  = idx / (kDM * kWW * kHH);
  const int r0 = (0 * kBB + bb) * kLSEQ + hh * kWW + ww;
  const int r1 = (1 * kBB + bb) * kLSEQ + hh * kWW + (kWW - 1 - ww);
  const int r2 = (2 * kBB + bb) * kLSEQ + ww * kHH + hh;
  const int r3 = (3 * kBB + bb) * kLSEQ + ww * kHH + (kHH - 1 - hh);
  const float sum = yp[(size_t)r0 * kDM + cch] + yp[(size_t)r1 * kDM + cch] +
                    yp[(size_t)r2 * kDM + cch] + yp[(size_t)r3 * kDM + cch];
  x1[idx] = x[idx] + 0.25f * sum;
}

// ---------------------------------------------------------------------------
// LayerNorm to bf16 (for MLP input).
__global__ void ln_bf16(const float* __restrict__ x,
                        const float* __restrict__ w,
                        const float* __restrict__ b,
                        bf16_t* __restrict__ o) {
  const int row = blockIdx.x;
  const int tid = threadIdx.x;
  __shared__ float red[256];
  const float* rp = x + (size_t)row * kDM;
  const float v = (tid < kDM) ? rp[tid] : 0.0f;
  red[tid] = v; __syncthreads();
  for (int s = 128; s > 0; s >>= 1) { if (tid < s) red[tid] += red[tid + s]; __syncthreads(); }
  const float mu = red[0] / kDM; __syncthreads();
  const float d = (tid < kDM) ? (v - mu) : 0.0f;
  red[tid] = d * d; __syncthreads();
  for (int s = 128; s > 0; s >>= 1) { if (tid < s) red[tid] += red[tid + s]; __syncthreads(); }
  const float var = red[0] / kDM;
  if (tid < kDM)
    o[(size_t)row * kDM + tid] =
        (bf16_t)((v - mu) * rsqrtf(var + 1e-5f) * w[tid] + b[tid]);
}

// f32 -> bf16 with zero-padding to rows_pad rows (for weight matrices).
__global__ void pack_w_bf16(const float* __restrict__ in, bf16_t* __restrict__ out,
                            int rows, int K, int rows_pad) {
  const int i = blockIdx.x * blockDim.x + threadIdx.x;
  if (i >= rows_pad * K) return;
  const int r = i / K;
  out[i] = (r < rows) ? (bf16_t)in[i] : (bf16_t)0.0f;
}

// ---------------------------------------------------------------------------
extern "C" void kernel_launch(void* const* d_in, const int* in_sizes, int n_in,
                              void* d_out, int out_size, void* d_ws, size_t ws_size,
                              hipStream_t stream) {
  (void)in_sizes; (void)n_in; (void)out_size; (void)ws_size;
  const float* x          = (const float*)d_in[0];
  const float* norm_w     = (const float*)d_in[1];
  const float* norm_b     = (const float*)d_in[2];
  const float* in_proj_w  = (const float*)d_in[3];
  const float* conv_w     = (const float*)d_in[4];
  const float* conv_b     = (const float*)d_in[5];
  const float* dt_bias    = (const float*)d_in[6];
  const float* A_log      = (const float*)d_in[7];
  const float* Dh         = (const float*)d_in[8];
  const float* rms_w      = (const float*)d_in[9];
  const float* out_proj_w = (const float*)d_in[10];
  const float* norm2_w    = (const float*)d_in[11];
  const float* norm2_b    = (const float*)d_in[12];
  const float* mlp_w1     = (const float*)d_in[13];
  const float* mlp_b1     = (const float*)d_in[14];
  const float* mlp_w2     = (const float*)d_in[15];
  const float* mlp_b2     = (const float*)d_in[16];
  float* out = (float*)d_out;

  constexpr int kDPROJp = pad32(kDPROJ);      // 1056
  uint8_t* ws = (uint8_t*)d_ws;
  size_t cur = 0;
  auto carve = [&](size_t bytes) -> void* {
    void* p = ws + cur;
    cur = (cur + bytes + 255) & ~(size_t)255;
    return p;
  };

  bf16_t* xs_bf    = (bf16_t*)carve((size_t)kMROWS * kDM * 2);
  bf16_t* w_inproj = (bf16_t*)carve((size_t)kDPROJp * kDM * 2);
  bf16_t* w_outprj = (bf16_t*)carve((size_t)kDM * kDIN * 2);
  bf16_t* w_mlp1   = (bf16_t*)carve((size_t)4 * kDM * kDM * 2);
  bf16_t* w_mlp2   = (bf16_t*)carve((size_t)4 * kDM * kDM * 2);
  float*  zx       = (float*) carve((size_t)kMROWS * kDPROJ * 4);
  float*  xh       = (float*) carve((size_t)kMROWS * kDIN * 4);
  bf16_t* Xb       = (bf16_t*)carve((size_t)kMROWS * kDIN * 2);
  bf16_t* Bb       = (bf16_t*)carve((size_t)kMROWS * 128 * 2);
  bf16_t* Cb       = (bf16_t*)carve((size_t)kMROWS * 128 * 2);
  float*  dA       = (float*) carve((size_t)kMROWS * kNH * 4);
  float*  Acum     = (float*) carve((size_t)kNSEQ * kNH * kLSEQ * 4);
  float*  cdec     = (float*) carve((size_t)kNSEQ * kNH * kNCHUNK * 4);
  float*  states   = (float*) carve((size_t)kNSEQ * kNH * kNCHUNK * 64 * 128 * 4);
  float*  Ybuf     = (float*) carve((size_t)kMROWS * kNH * kHD * 4);
  bf16_t* yg       = (bf16_t*)carve((size_t)kMROWS * kDIN * 2);
  float*  yproj    = (float*) carve((size_t)kMROWS * kDM * 4);
  float*  x1       = (float*) carve((size_t)kPIX * kDM * 4);
  bf16_t* xn2      = (bf16_t*)carve((size_t)kPIX * kDM * 2);
  bf16_t* hbuf     = (bf16_t*)carve((size_t)kPIX * 4 * kDM * 2);

  // Weight conversions to bf16 (zero-padded rows)
  pack_w_bf16<<<(kDPROJp * kDM + 255) / 256, 256, 0, stream>>>(
      in_proj_w, w_inproj, kDPROJ, kDM, kDPROJp);
  pack_w_bf16<<<(kDM * kDIN + 255) / 256, 256, 0, stream>>>(
      out_proj_w, w_outprj, kDM, kDIN, kDM);
  pack_w_bf16<<<(4 * kDM * kDM + 255) / 256, 256, 0, stream>>>(
      mlp_w1, w_mlp1, 4 * kDM, kDM, 4 * kDM);
  pack_w_bf16<<<(4 * kDM * kDM + 255) / 256, 256, 0, stream>>>(
      mlp_w2, w_mlp2, kDM, 4 * kDM, kDM);

  // 1) LayerNorm + cross-scan
  ln_crossscan<<<kPIX, 256, 0, stream>>>(x, norm_w, norm_b, xs_bf);

  // 2) in_proj GEMM (32768 x 1030, K=192)
  {
    const int pairs = (((kDPROJ + 15) / 16) + 1) / 2;
    const int tiles = (kMROWS / 16) * pairs;
    gemm_wmma_bf16<<<(tiles + 7) / 8, 256, 0, stream>>>(
        xs_bf, w_inproj, nullptr, nullptr, zx, nullptr, kMROWS, kDPROJ, kDM, 0);
  }

  // 3) conv + gates
  conv_gate<<<kMROWS, 256, 0, stream>>>(zx, conv_w, conv_b, dt_bias, A_log,
                                        xh, Xb, Bb, Cb, dA);

  // 4-6) SSD
  dim3 g3(kNCHUNK, kNH, kNSEQ);
  ssd_intra<<<g3, 256, 0, stream>>>(Bb, Cb, Xb, dA, Acum, cdec, states, Ybuf);
  ssd_scan<<<kNSEQ * kNH, 256, 0, stream>>>(states, cdec);
  ssd_inter<<<g3, 256, 0, stream>>>(Cb, Acum, states, Ybuf);

  // 7) gate + rmsnorm
  gate_rms<<<kMROWS, 128, 0, stream>>>(Ybuf, xh, zx, Dh, rms_w, yg);

  // 8) out_proj GEMM (32768 x 192, K=384)
  {
    const int pairs = ((kDM / 16) + 1) / 2;
    const int tiles = (kMROWS / 16) * pairs;
    gemm_wmma_bf16<<<(tiles + 7) / 8, 256, 0, stream>>>(
        yg, w_outprj, nullptr, nullptr, yproj, nullptr, kMROWS, kDM, kDIN, 0);
  }

  // 9) cross-merge + residual
  cross_merge<<<(kPIX * kDM + 255) / 256, 256, 0, stream>>>(x, yproj, x1);

  // 10) LN2 -> bf16
  ln_bf16<<<kPIX, 256, 0, stream>>>(x1, norm2_w, norm2_b, xn2);

  // 11) MLP1 (8192 x 768, K=192) + bias + GELU, bf16 out
  {
    const int pairs = ((4 * kDM / 16) + 1) / 2;
    const int tiles = (kPIX / 16) * pairs;
    gemm_wmma_bf16<<<(tiles + 7) / 8, 256, 0, stream>>>(
        xn2, w_mlp1, mlp_b1, nullptr, nullptr, hbuf, kPIX, 4 * kDM, kDM,
        EPI_BIAS | EPI_GELU | EPI_BF16);
  }

  // 12) MLP2 (8192 x 192, K=768) + bias + residual -> d_out
  {
    const int pairs = ((kDM / 16) + 1) / 2;
    const int tiles = (kPIX / 16) * pairs;
    gemm_wmma_bf16<<<(tiles + 7) / 8, 256, 0, stream>>>(
        hbuf, w_mlp2, mlp_b2, x1, out, nullptr, kPIX, kDM, 4 * kDM,
        EPI_BIAS | EPI_RESID);
  }
}
